// VectorQuantizerEMA_29497835389284
// MI455X (gfx1250) — compile-verified, matrix operand order checked by reading, not run on hardware
//
#include <hip/hip_runtime.h>

// ---------------------------------------------------------------------------
// VQ nearest-codebook search on MI455X (gfx1250, wave32, WMMA).
//   score(n,k) = dot(z_n, e_k) - 0.5*||e_k||^2   (argmax == reference argmin)
// Split-f16 dot (hi/lo, drop lo*lo): 3 WMMAs per 32-K chunk, f32 accumulate
// -> ~f32-exact argmin at 1/3 the matrix-op count of the f32 WMMA path.
// v2 changes vs v1 (driven by disasm):
//   * M=32 per wave: each B fragment (LDS) now feeds 6 WMMAs instead of 3
//     (24 WMMA : 16 ds_load_b128 per code-tile) -- halves LDS bytes/FLOP,
//     which was the binding resource at M=16.
//   * Both accumulator chains start from inline C=0; -0.5||e||^2 added in the
//     epilogue (removes the 7-mov broadcast + ds-load dependency per tile).
//   * Pointer-walk LDS addressing (+4352 B/tile) -> ds-offset immediates.
//   * 4 line-granular prefetches per wave cover the entire next A tile
//     (32 rows x 4B = one 128B line per channel) instead of 128 per lane.
//   * A loads / C stores use one base register + compile-time offsets;
//     output uses non-temporal stores (streamed, never re-read).
// ---------------------------------------------------------------------------

typedef __attribute__((ext_vector_type(16))) _Float16 v16h;
typedef __attribute__((ext_vector_type(8)))  _Float16 v8h;
typedef __attribute__((ext_vector_type(2)))  _Float16 v2h;
typedef __attribute__((ext_vector_type(8)))  float    v8f;

#define K_CODES   512
#define CDIM      128
#define HWSZ      4096                  // 64*64
#define NROWS     (32 * HWSZ)           // 131072 query vectors
#define DTILES    (NROWS / 32)          // 4096 double row-tiles (M=32 each)

#define LDS_ROW   136                   // halfs per code row (68-dword stride: conflict-free)
#define EL_BYTE   (K_CODES * LDS_ROW * 2)          // byte offset of lo array
#define ESQ_BYTE  (2 * EL_BYTE)                    // byte offset of esq
#define SMEM_BYTES (ESQ_BYTE + K_CODES * 4)        // 280576 B of 320KB/WGP

#define WMMA_F16(A, B, C) \
  __builtin_amdgcn_wmma_f32_16x16x32_f16(false, (A), false, (B), (short)0, (C), false, false)

__global__ __launch_bounds__(256, 1) void
vq_argmin_wmma(const float* __restrict__ z, const float* __restrict__ emb,
               float* __restrict__ out) {
  extern __shared__ char smem[];
  _Float16* eh  = (_Float16*)smem;
  _Float16* el  = (_Float16*)(smem + EL_BYTE);
  float*    esq = (float*)(smem + ESQ_BYTE);

  const int tid = threadIdx.x;

  // ---- Stage codebook into LDS as split f16 (hi + lo), padded rows ----
  for (int i = tid; i < K_CODES * (CDIM / 2); i += blockDim.x) {
    const int code = i >> 6;
    const int cp   = (i & 63) * 2;
    const float x0 = emb[code * CDIM + cp];
    const float x1 = emb[code * CDIM + cp + 1];
    const _Float16 h0 = (_Float16)x0, h1 = (_Float16)x1;
    const _Float16 l0 = (_Float16)(x0 - (float)h0);
    const _Float16 l1 = (_Float16)(x1 - (float)h1);
    const int base = code * LDS_ROW + cp;
    *(v2h*)&eh[base] = (v2h){h0, h1};
    *(v2h*)&el[base] = (v2h){l0, l1};
  }
  for (int code = tid; code < K_CODES; code += blockDim.x) {
    float s = 0.f;
    for (int c = 0; c < CDIM; ++c) { const float v = emb[code * CDIM + c]; s += v * v; }
    esq[code] = -0.5f * s;
  }
  __syncthreads();

  const int lane  = tid & 31;
  const int half  = lane >> 4;            // 0/1: which 16-lane half
  const int l15   = lane & 15;
  const int off   = half * 8;             // K offset of this half within a 32-chunk
  const int wave  = tid >> 5;
  const int gwave = blockIdx.x * (blockDim.x >> 5) + wave;
  const int nwav  = gridDim.x * (blockDim.x >> 5);

  for (int dt = gwave; dt < DTILES; dt += nwav) {
    const int    m0  = dt * 32;           // 32 rows per wave, never crosses a batch (4096|32)
    const size_t zb0 = (size_t)(m0 >> 12) * CDIM * HWSZ + (m0 & (HWSZ - 1)) + l15;
    const float* zp  = z + zb0 + (size_t)off * HWSZ;   // all A offsets now compile-time

    // ---- prefetch next double-tile: one 128B line per channel, 4 instrs ----
    const int ndt = dt + nwav;
    if (ndt < DTILES) {
      const int nm0 = ndt * 32;
      const float* zpn = z + (size_t)(nm0 >> 12) * CDIM * HWSZ + (nm0 & (HWSZ - 1));
#pragma unroll
      for (int i = 0; i < 4; ++i)
        __builtin_prefetch(&zpn[(size_t)(lane + 32 * i) * HWSZ], 0, 1);
    }

    // ---- Load A (32 rows x 128 C), split into f16 hi/lo register fragments ----
    v16h ah[2][4], al[2][4];
#pragma unroll
    for (int g = 0; g < 2; ++g) {
#pragma unroll
      for (int q = 0; q < 4; ++q) {
        float xs[16];
#pragma unroll
        for (int j = 0; j < 8; ++j) {
          xs[j]     = zp[(size_t)(q * 32 + j)      * HWSZ + g * 16];
          xs[8 + j] = zp[(size_t)(q * 32 + 16 + j) * HWSZ + g * 16];
        }
#pragma unroll
        for (int e = 0; e < 16; ++e) {
          const _Float16 h = (_Float16)xs[e];
          ah[g][q][e] = h;
          al[g][q][e] = (_Float16)(xs[e] - (float)h);
        }
      }
    }

    float best[2][8];
    int   bidx[2][8];
#pragma unroll
    for (int g = 0; g < 2; ++g)
#pragma unroll
      for (int v = 0; v < 8; ++v) { best[g][v] = -3.4e38f; bidx[g][v] = 0; }

    // ---- Sweep all 512 codes, 16 at a time; pointer-walk LDS addressing ----
    const char*  peh  = (const char*)smem + (size_t)(l15 * LDS_ROW + off) * 2;
    const char*  pel  = peh + EL_BYTE;
    const float* pesq = esq + l15;

    for (int t = 0; t < K_CODES / 16; ++t) {
      const float cinit = *pesq;          // -0.5*||e_n||^2, n = t*16 + l15
      v8f acc0h = {}, acc0l = {}, acc1h = {}, acc1l = {};   // inline C=0 starts
#pragma unroll
      for (int q = 0; q < 4; ++q) {
        const v8h h0 = *(const v8h*)(peh + q * 64);
        const v8h h1 = *(const v8h*)(peh + q * 64 + 32);
        const v8h l0 = *(const v8h*)(pel + q * 64);
        const v8h l1 = *(const v8h*)(pel + q * 64 + 32);
        const v16h bh = __builtin_shufflevector(h0, h1, 0,1,2,3,4,5,6,7,8,9,10,11,12,13,14,15);
        const v16h bl = __builtin_shufflevector(l0, l1, 0,1,2,3,4,5,6,7,8,9,10,11,12,13,14,15);
        acc0h = WMMA_F16(ah[0][q], bh, acc0h);   // each B fragment feeds 6 WMMAs
        acc1h = WMMA_F16(ah[1][q], bh, acc1h);
        acc0l = WMMA_F16(ah[0][q], bl, acc0l);
        acc1l = WMMA_F16(ah[1][q], bl, acc1l);
        acc0l = WMMA_F16(al[0][q], bh, acc0l);
        acc1l = WMMA_F16(al[1][q], bh, acc1l);
      }
      const int n = t * 16 + l15;
#pragma unroll
      for (int v = 0; v < 8; ++v) {
        const float s0 = acc0h[v] + acc0l[v] + cinit;
        const float s1 = acc1h[v] + acc1l[v] + cinit;
        if (s0 > best[0][v]) { best[0][v] = s0; bidx[0][v] = n; }  // strict >: lowest n on tie
        if (s1 > best[1][v]) { best[1][v] = s1; bidx[1][v] = n; }
      }
      peh  += 16 * LDS_ROW * 2;
      pel  += 16 * LDS_ROW * 2;
      pesq += 16;
    }

    // ---- Cross-lane argmax within each 16-lane half (rows 0-7 / 8-15) ----
#pragma unroll
    for (int g = 0; g < 2; ++g)
#pragma unroll
      for (int v = 0; v < 8; ++v)
#pragma unroll
        for (int m = 1; m < 16; m <<= 1) {
          const float ob = __shfl_xor(best[g][v], m, 32);
          const int   oi = __shfl_xor(bidx[g][v], m, 32);
          if (ob > best[g][v] || (ob == best[g][v] && oi < bidx[g][v])) {
            best[g][v] = ob; bidx[g][v] = oi;
          }
        }

    // swap halves so every lane can pick the winner for row (lane & 15)
    int sel[2];
#pragma unroll
    for (int g = 0; g < 2; ++g) {
      int oidx[8], rowsLo[8], rowsHi[8];
#pragma unroll
      for (int v = 0; v < 8; ++v) oidx[v] = __shfl_xor(bidx[g][v], 16, 32);
#pragma unroll
      for (int v = 0; v < 8; ++v) {
        rowsLo[v] = half ? oidx[v] : bidx[g][v];   // rows 0..7
        rowsHi[v] = half ? bidx[g][v] : oidx[v];   // rows 8..15
      }
      int s = (l15 < 8) ? rowsLo[0] : rowsHi[0];
#pragma unroll
      for (int v = 1; v < 8; ++v)
        if ((l15 & 7) == v) s = (l15 < 8) ? rowsLo[v] : rowsHi[v];
      sel[g] = s;
    }

    // ---- Gather winning codes (exact f32, L2-resident) and stream out ----
    float* op = out + zb0 + (size_t)half * HWSZ;     // channel base incl. half
    const float* e0 = emb + sel[0] * CDIM + half;
    const float* e1 = emb + sel[1] * CDIM + half;
#pragma unroll 4
    for (int cc = 0; cc < CDIM; cc += 2) {
      __builtin_nontemporal_store(e0[cc], &op[(size_t)cc * HWSZ]);       // rows m0..m0+15
      __builtin_nontemporal_store(e1[cc], &op[(size_t)cc * HWSZ + 16]);  // rows m0+16..+31
    }
  }
}

extern "C" void kernel_launch(void* const* d_in, const int* in_sizes, int n_in,
                              void* d_out, int out_size, void* d_ws, size_t ws_size,
                              hipStream_t stream) {
  const float* z   = (const float*)d_in[0];   // [32,128,64,64] f32
  const float* emb = (const float*)d_in[1];   // [512,128] f32
  float*       out = (float*)d_out;           // [32,128,64,64] f32
  (void)in_sizes; (void)n_in; (void)out_size; (void)d_ws; (void)ws_size;

  // 256 threads = 8 waves/WG, one WG per WGP (280KB LDS). 4096 double-tiles,
  // 2048 waves -> exactly 2 per wave, zero divergence at barriers/loops.
  dim3 grid(256), block(256);
  vq_argmin_wmma<<<grid, block, SMEM_BYTES, stream>>>(z, emb, out);
}
